// QuantLlamaAttention_87986700025856
// MI455X (gfx1250) — compile-verified, matrix operand order checked by reading, not run on hardware
//
#include <hip/hip_runtime.h>
#include <hip/hip_bf16.h>
#include <math.h>

// ---------------------------------------------------------------------------
// Problem constants (from reference)
// ---------------------------------------------------------------------------
#define S_LEN   2000
#define HDIM    4096
#define NH      32
#define NKV     8
#define HD      128
#define NREP    (NH / NKV)          // 4
#define QS_LEN  1920                // S - S % RES_LEN (residual = 80 FP tokens)
#define NEGBIG  (-1000000000.0f)

typedef __attribute__((ext_vector_type(16))) __bf16          v16bf;
typedef __attribute__((ext_vector_type(16))) unsigned short  v16us;
typedef __attribute__((ext_vector_type(8)))  unsigned short  v8us;
typedef __attribute__((ext_vector_type(8)))  float           v8f;
typedef __attribute__((ext_vector_type(4)))  int             v4i;

// fp32 -> bf16 (round-to-nearest-even) as raw bits
static __device__ __forceinline__ unsigned short f2bf(float f) {
    unsigned int u = __float_as_uint(f);
    unsigned int r = u + 0x7FFFu + ((u >> 16) & 1u);
    return (unsigned short)(r >> 16);
}
static __device__ __forceinline__ unsigned int pack2(float lo, float hi) {
#if __has_builtin(__builtin_amdgcn_cvt_pk_bf16_f32)
    typedef __attribute__((ext_vector_type(2))) __bf16 v2bf;
    v2bf r = __builtin_amdgcn_cvt_pk_bf16_f32(lo, hi);
    return __builtin_bit_cast(unsigned int, r);
#else
    return (unsigned int)f2bf(lo) | ((unsigned int)f2bf(hi) << 16);
#endif
}

// ---------------------------------------------------------------------------
// Async global->LDS copies (gfx1250 GLOBAL_LOAD_ASYNC_TO_LDS_B128, ASYNCcnt)
// with VGPR-copy fallback if toolchain lacks the builtin.
// Probe-learned signature: (v4i AS1*, v4i AS3*, imm offset, imm cpol).
// ---------------------------------------------------------------------------
#if __has_builtin(__builtin_amdgcn_global_load_async_to_lds_b128) && \
    __has_builtin(__builtin_amdgcn_s_wait_asynccnt)
#define HAS_ASYNC_LDS 1
#else
#define HAS_ASYNC_LDS 0
#endif

static __device__ __forceinline__ void cp_b128(unsigned short* ldst,
                                               const unsigned short* gsrc) {
#if HAS_ASYNC_LDS
    __builtin_amdgcn_global_load_async_to_lds_b128(
        (__attribute__((address_space(1))) v4i*)gsrc,
        (__attribute__((address_space(3))) v4i*)ldst, 0, 0);
#else
    *(v8us*)ldst = *(const v8us*)gsrc;
#endif
}
static __device__ __forceinline__ void wait_async() {
#if HAS_ASYNC_LDS
    __builtin_amdgcn_s_wait_asynccnt(0);
#endif
}

// Operand fetch: lane's 16 bf16 = [k .. k+7] and [k+16 .. k+23] of one LDS row,
// both 16B-aligned -> two ds_load_b128.
static __device__ __forceinline__ v16us ld_op(const unsigned short* p) {
    v8us lo = *(const v8us*)p;
    v8us hi = *(const v8us*)(p + 16);
    return __builtin_shufflevector(lo, hi, 0, 1, 2, 3, 4, 5, 6, 7,
                                           8, 9, 10, 11, 12, 13, 14, 15);
}
// Half operand (K padded 16->32 with zeros): one b128 load + zero upper half.
static __device__ __forceinline__ v16us ld_op_half(const unsigned short* p) {
    v8us lo = *(const v8us*)p;
    v16us r;
#pragma unroll
    for (int e = 0; e < 8; ++e) { r[e] = lo[e]; r[e + 8] = 0; }
    return r;
}

#define WMMA_BF16(A, B, C)                                                     \
    __builtin_amdgcn_wmma_f32_16x16x32_bf16(                                   \
        false, __builtin_bit_cast(v16bf, (A)),                                 \
        false, __builtin_bit_cast(v16bf, (B)),                                 \
        (short)0, (C), false, false)

// ---------------------------------------------------------------------------
// Conversions (one pass per launch; removes fp32 streams from hot loops)
// ---------------------------------------------------------------------------
__global__ void conv_bf16_kernel(const float* __restrict__ src,
                                 unsigned short* __restrict__ dst, int n2)
{
    int i = blockIdx.x * blockDim.x + threadIdx.x;
    if (i >= n2) return;
    float2 f = *(const float2*)(src + 2 * (size_t)i);
    *(unsigned int*)(dst + 2 * (size_t)i) = pack2(f.x, f.y);
}

// W[K][N] f32 -> WT[N][K] bf16 via 32x32 LDS tile (coalesced both sides)
__global__ __launch_bounds__(256) void convT_kernel(
    const float* __restrict__ W, unsigned short* __restrict__ WT, int K, int N)
{
    __shared__ float t[32][33];
    const int tid = threadIdx.x;
    const int k0 = blockIdx.y * 32, n0 = blockIdx.x * 32;
#pragma unroll
    for (int it = 0; it < 4; ++it) {
        int i = it * 256 + tid;
        int kk = i >> 5, nn = i & 31;
        t[kk][nn] = W[(size_t)(k0 + kk) * N + n0 + nn];
    }
    __syncthreads();
#pragma unroll
    for (int it = 0; it < 2; ++it) {
        int i = it * 256 + tid;
        int nn = i >> 4, kp = (i & 15) * 2;
        *(unsigned int*)&WT[(size_t)(n0 + nn) * K + k0 + kp] =
            pack2(t[kp][nn], t[kp + 1][nn]);
    }
}

// q/k f32 [s][nh*128] -> bf16 [h][s][128]
__global__ void conv_head_kernel(const float* __restrict__ src,
                                 unsigned short* __restrict__ dst,
                                 int nh, int total2)
{
    int i = blockIdx.x * blockDim.x + threadIdx.x;
    if (i >= total2) return;
    int d2 = i & 63;
    int s  = (i >> 6) % S_LEN;
    int h  = i / (64 * S_LEN);
    float2 f = *(const float2*)(src + (size_t)s * nh * HD + h * HD + 2 * d2);
    *(unsigned int*)(dst + ((size_t)h * S_LEN + s) * HD + 2 * d2) = pack2(f.x, f.y);
}

// v f32 [s][NKV*128] -> bf16 transposed [kvh][d][s] (pairs along s)
__global__ void conv_vT_kernel(const float* __restrict__ src,
                               unsigned short* __restrict__ dst)
{
    int i = blockIdx.x * blockDim.x + threadIdx.x;
    const int SP = S_LEN / 2;
    if (i >= NKV * HD * SP) return;
    int s2  = i % SP;
    int d   = (i / SP) % HD;
    int kvh = i / (HD * SP);
    int s0  = 2 * s2;
    float f0 = src[(size_t)s0 * (NKV * HD) + kvh * HD + d];
    float f1 = src[(size_t)(s0 + 1) * (NKV * HD) + kvh * HD + d];
    *(unsigned int*)(dst + ((size_t)kvh * HD + d) * S_LEN + s0) = pack2(f0, f1);
}

// ---------------------------------------------------------------------------
// Kernel 1: C[M,N] = A[M,K] * B[K,N], bf16 in (B pre-transposed [N][K]),
// f32 out. 256 threads = 8 waves, tile 64(M) x 128(N) x 32(K).
// Double-buffered LDS: async-stage tile t+1 while computing tile t.
// ---------------------------------------------------------------------------
__global__ __launch_bounds__(256) void gemm_bf16_kernel(
    const unsigned short* __restrict__ A, const unsigned short* __restrict__ BT,
    float* __restrict__ C, int M, int N, int K)
{
    __shared__ unsigned short As[2][64][40];    // [buf][m][k]
    __shared__ unsigned short BsT[2][128][40];  // [buf][n][k]

    const int tid  = threadIdx.x;
    const int lane = tid & 31;
    const int w    = tid >> 5;
    const int mi   = w >> 1;                 // 0..3
    const int ni   = w & 1;                  // 0..1
    const int m0   = blockIdx.y * 64;
    const int n0   = blockIdx.x * 128;

    const int rA    = lane & 15;
    const int kbase = (lane >> 4) ? 8 : 0;
    const int mloc  = kbase;

    // Staging coordinates
    const int ar = tid >> 2, ac = (tid & 3) * 8;
    int gra = m0 + ar; if (gra >= M) gra = M - 1;   // clamp: rows m>=M unused

    v8f acc[4];
#pragma unroll
    for (int c = 0; c < 4; ++c)
#pragma unroll
        for (int e = 0; e < 8; ++e) acc[c][e] = 0.0f;

    const int nt = K / 32;

    // Prologue: stage tile 0 into buffer 0
    cp_b128(&As[0][ar][ac], A + (size_t)gra * K + 0 + ac);
#pragma unroll
    for (int ii = 0; ii < 2; ++ii) {
        int i = tid + ii * 256;
        int n = i >> 2, c8 = (i & 3) * 8;
        cp_b128(&BsT[0][n][c8], BT + (size_t)(n0 + n) * K + 0 + c8);
    }
    wait_async();
    __syncthreads();

    for (int t = 0; t < nt; ++t) {
        const int cur = t & 1, nxt = cur ^ 1;
        // Stage tile t+1 into the other buffer (overlaps with compute)
        if (t + 1 < nt) {
            int k1 = 32 * (t + 1);
            cp_b128(&As[nxt][ar][ac], A + (size_t)gra * K + k1 + ac);
#pragma unroll
            for (int ii = 0; ii < 2; ++ii) {
                int i = tid + ii * 256;
                int n = i >> 2, c8 = (i & 3) * 8;
                cp_b128(&BsT[nxt][n][c8], BT + (size_t)(n0 + n) * K + k1 + c8);
            }
        }
        if (t + 2 < nt) {
            __builtin_prefetch(&A[(size_t)gra * K + 32 * (t + 2)], 0, 1);
            __builtin_prefetch(&BT[(size_t)(n0 + (tid >> 1)) * K + 32 * (t + 2)], 0, 1);
        }

        // Compute tile t from current buffer
        v16us a = ld_op(&As[cur][16 * mi + rA][kbase]);
#pragma unroll
        for (int c = 0; c < 4; ++c) {
            v16us b = ld_op(&BsT[cur][64 * ni + 16 * c + (lane & 15)][kbase]);
            acc[c] = WMMA_BF16(a, b, acc[c]);
        }

        wait_async();
        __syncthreads();
    }

#pragma unroll
    for (int c = 0; c < 4; ++c)
#pragma unroll
        for (int r = 0; r < 8; ++r) {
            int m = m0 + 16 * mi + r + mloc;
            int n = n0 + 64 * ni + 16 * c + (lane & 15);
            if (m < M) C[(size_t)m * N + n] = acc[c][r];
        }
}

// ---------------------------------------------------------------------------
// Kernel 2: RoPE in place on x[S][nh*128]; pair (d, d+64), freq index d.
// ---------------------------------------------------------------------------
__global__ void rope_kernel(float* __restrict__ x,
                            const long long* __restrict__ pos,
                            int nh, int total)
{
    int idx = blockIdx.x * blockDim.x + threadIdx.x;
    if (idx >= total) return;
    int d = idx % 64;
    int h = (idx / 64) % nh;
    int s = idx / (64 * nh);
    float p    = (float)pos[s];
    float freq = powf(10000.0f, -(2.0f * (float)d) / 128.0f);
    float ang  = p * freq;
    float cs = cosf(ang), sn = sinf(ang);
    float* row = x + (size_t)s * nh * HD + h * HD;
    float x1 = row[d], x2 = row[d + 64];
    row[d]      = x1 * cs - x2 * sn;
    row[d + 64] = x2 * cs + x1 * sn;
}

// ---------------------------------------------------------------------------
// Kernel 3: per-(kvh,dim) stats over sequence: scale = max|k|^0.6, bias = mean
// ---------------------------------------------------------------------------
__global__ __launch_bounds__(256) void kstat_kernel(
    const float* __restrict__ k, float* __restrict__ scale,
    float* __restrict__ bias)
{
    int hd = blockIdx.x;
    __shared__ float smax[256];
    __shared__ float ssum[256];
    float mx = 0.0f, sm = 0.0f;
    for (int s = threadIdx.x; s < S_LEN; s += 256) {
        float v = k[(size_t)s * (NKV * HD) + hd];
        mx = fmaxf(mx, fabsf(v));
        sm += v;
    }
    smax[threadIdx.x] = mx;
    ssum[threadIdx.x] = sm;
    __syncthreads();
    for (int off = 128; off > 0; off >>= 1) {
        if (threadIdx.x < off) {
            smax[threadIdx.x] = fmaxf(smax[threadIdx.x], smax[threadIdx.x + off]);
            ssum[threadIdx.x] += ssum[threadIdx.x + off];
        }
        __syncthreads();
    }
    if (threadIdx.x == 0) {
        scale[hd] = powf(smax[0], 0.6f);
        bias[hd]  = ssum[0] / (float)S_LEN;
    }
}

// ---------------------------------------------------------------------------
// Kernel 4a/4b: 4-bit group-64 quant-dequant in place, tokens [0, QS_LEN).
// ---------------------------------------------------------------------------
__global__ void quantk_kernel(float* __restrict__ k,
                              const float* __restrict__ scale,
                              const float* __restrict__ bias)
{
    int g    = (blockIdx.x * blockDim.x + threadIdx.x) >> 5;
    int lane = threadIdx.x & 31;
    if (g >= QS_LEN * NKV * 2) return;
    int s    = g >> 4;
    int rem  = g & 15;
    int kvh  = rem >> 1;
    int half = rem & 1;
    int d0   = kvh * HD + half * 64 + 2 * lane;
    size_t base = (size_t)s * (NKV * HD) + d0;

    float v0 = k[base], v1 = k[base + 1];
    float sc0 = scale[d0], sc1 = scale[d0 + 1];
    float b0  = bias[d0],  b1  = bias[d0 + 1];
    float x0 = (v0 - b0) / sc0, x1 = (v1 - b1) / sc1;

    float mx = fmaxf(x0, x1), mn = fminf(x0, x1);
    for (int m = 1; m < 32; m <<= 1) {
        mx = fmaxf(mx, __shfl_xor(mx, m, 32));
        mn = fminf(mn, __shfl_xor(mn, m, 32));
    }
    float gs = fmaxf((mx - mn) / 15.0f, 1e-8f);
    float q0 = fminf(fmaxf(rintf((x0 - mn) / gs), 0.0f), 15.0f);
    float q1 = fminf(fmaxf(rintf((x1 - mn) / gs), 0.0f), 15.0f);
    k[base]     = (q0 * gs + mn) * sc0 + b0;
    k[base + 1] = (q1 * gs + mn) * sc1 + b1;
}

__global__ void quantv_kernel(float* __restrict__ v)
{
    int g    = (blockIdx.x * blockDim.x + threadIdx.x) >> 5;
    int lane = threadIdx.x & 31;
    if (g >= QS_LEN * NKV * 2) return;
    int s    = g >> 4;
    int rem  = g & 15;
    int kvh  = rem >> 1;
    int half = rem & 1;
    size_t base = (size_t)s * (NKV * HD) + kvh * HD + half * 64 + 2 * lane;

    float x0 = v[base], x1 = v[base + 1];
    float mx = fmaxf(x0, x1), mn = fminf(x0, x1);
    for (int m = 1; m < 32; m <<= 1) {
        mx = fmaxf(mx, __shfl_xor(mx, m, 32));
        mn = fminf(mn, __shfl_xor(mn, m, 32));
    }
    float gs = fmaxf((mx - mn) / 15.0f, 1e-8f);
    float q0 = fminf(fmaxf(rintf((x0 - mn) / gs), 0.0f), 15.0f);
    float q1 = fminf(fmaxf(rintf((x1 - mn) / gs), 0.0f), 15.0f);
    v[base]     = q0 * gs + mn;
    v[base + 1] = q1 * gs + mn;
}

// ---------------------------------------------------------------------------
// Kernel 5: flash attention. 5 waves/WG handle 5 consecutive 16-row Q blocks
// of one head, sharing double-buffered K/V tiles staged asynchronously.
// QK^T: 4 WMMA; P@V: 8 WMMA. Fully-masked blocks contribute exp(-1e9-m)=0.
// ---------------------------------------------------------------------------
__global__ __launch_bounds__(160) void attn_kernel(
    const unsigned short* __restrict__ qbf,   // [NH][S][HD]
    const unsigned short* __restrict__ kbf,   // [NKV][S][HD]
    const unsigned short* __restrict__ vbfT,  // [NKV][HD][S]
    unsigned short* __restrict__ ctxbf)       // [S][NH*HD]
{
    const int wg   = blockIdx.x;            // 0..24
    const int h    = blockIdx.y;            // 0..NH-1
    const int kvh  = h >> 2;
    const int tid  = threadIdx.x;
    const int w    = tid >> 5;              // wave 0..4
    const int lane = tid & 31;
    const int qb   = wg * 5 + w;
    const int q0   = qb * 16;
    const int kbmax = wg * 5 + 4;

    __shared__ unsigned short Qs[5][16][136];
    __shared__ unsigned short Ks[2][16][136];
    __shared__ unsigned short VsT[2][128][24];
    __shared__ unsigned short Ps[5][16][24];

    const unsigned short* kbase_p = kbf + (size_t)kvh * S_LEN * HD;
    const unsigned short* vbase_p = vbfT + (size_t)kvh * HD * S_LEN;

    // Stage this wave's Q tile + K/V tile 0 (all async), one wait+barrier
    const unsigned short* qsrc = qbf + ((size_t)h * S_LEN + q0) * HD;
#pragma unroll
    for (int it = 0; it < 8; ++it) {
        int g  = it * 32 + lane;
        int r  = g >> 4;
        int c8 = (g & 15) * 8;
        cp_b128(&Qs[w][r][c8], qsrc + (size_t)r * HD + c8);
    }
    for (int i = tid; i < 256; i += 160)
        cp_b128(&Ks[0][i >> 4][(i & 15) * 8], kbase_p + i * 8);
    for (int i = tid; i < 256; i += 160) {
        int d = i >> 1, hf = i & 1;
        cp_b128(&VsT[0][d][hf * 8], vbase_p + (size_t)d * S_LEN + hf * 8);
    }
    wait_async();
    __syncthreads();

    const int row   = lane & 15;
    const int kb0   = (lane >> 4) ? 8 : 0;
    const int mloc  = kb0;
    const int n     = lane & 15;

    v16us aQ[4];
#pragma unroll
    for (int j = 0; j < 4; ++j)
        aQ[j] = ld_op(&Qs[w][row][32 * j + kb0]);

    v8f oacc[8];
#pragma unroll
    for (int c = 0; c < 8; ++c)
#pragma unroll
        for (int e = 0; e < 8; ++e) oacc[c][e] = 0.0f;
    float mrow[8], lrow[8];
#pragma unroll
    for (int r = 0; r < 8; ++r) { mrow[r] = -3.0e38f; lrow[r] = 0.0f; }

    const float inv_sqrt_d = 0.08838834764831845f;  // 1/sqrt(128)

    for (int kb = 0; kb <= kbmax; ++kb) {
        const int cur = kb & 1, nxt = cur ^ 1;
        // Stage K/V tile kb+1 into the other buffer (overlaps with compute)
        if (kb + 1 <= kbmax) {
            const unsigned short* ksrc = kbase_p + (size_t)(kb + 1) * 16 * HD;
            for (int i = tid; i < 256; i += 160)
                cp_b128(&Ks[nxt][i >> 4][(i & 15) * 8], ksrc + i * 8);
            const unsigned short* vsrc = vbase_p + (kb + 1) * 16;
            for (int i = tid; i < 256; i += 160) {
                int d = i >> 1, hf = i & 1;
                cp_b128(&VsT[nxt][d][hf * 8], vsrc + (size_t)d * S_LEN + hf * 8);
            }
        }

        // S = Q @ K^T
        v8f sacc;
#pragma unroll
        for (int e = 0; e < 8; ++e) sacc[e] = 0.0f;
#pragma unroll
        for (int j = 0; j < 4; ++j) {
            v16us bK = ld_op(&Ks[cur][n][32 * j + kb0]);
            sacc = WMMA_BF16(aQ[j], bK, sacc);
        }

        // mask + online softmax
        float pvals[8], cfv[8];
        const int kg = kb * 16 + n;
#pragma unroll
        for (int r = 0; r < 8; ++r) {
            int qg = q0 + r + mloc;
            float sv = sacc[r] * inv_sqrt_d;
            if (kg > qg) sv = NEGBIG;
            float rm = sv;
            rm = fmaxf(rm, __shfl_xor(rm, 1, 16));
            rm = fmaxf(rm, __shfl_xor(rm, 2, 16));
            rm = fmaxf(rm, __shfl_xor(rm, 4, 16));
            rm = fmaxf(rm, __shfl_xor(rm, 8, 16));
            float mnew = fmaxf(mrow[r], rm);
            float cf   = __expf(mrow[r] - mnew);
            float p    = __expf(sv - mnew);
            float rs = p;
            rs += __shfl_xor(rs, 1, 16);
            rs += __shfl_xor(rs, 2, 16);
            rs += __shfl_xor(rs, 4, 16);
            rs += __shfl_xor(rs, 8, 16);
            lrow[r] = lrow[r] * cf + rs;
            mrow[r] = mnew;
            pvals[r] = p;
            cfv[r]   = cf;
        }
#pragma unroll
        for (int c = 0; c < 8; ++c)
#pragma unroll
            for (int r = 0; r < 8; ++r) oacc[c][r] *= cfv[r];

        // Re-layout P (C layout -> A-operand layout) through per-wave LDS
#pragma unroll
        for (int r = 0; r < 8; ++r)
            Ps[w][r + mloc][n] = f2bf(pvals[r]);

        v16us aP = ld_op_half(&Ps[w][row][kb0]);
#pragma unroll
        for (int c = 0; c < 8; ++c) {
            v16us bV = ld_op_half(&VsT[cur][16 * c + n][kb0]);
            oacc[c] = WMMA_BF16(aP, bV, oacc[c]);
        }

        wait_async();
        __syncthreads();
    }

    // Epilogue: normalize, write bf16 ctx[S][NH*HD]
#pragma unroll
    for (int c = 0; c < 8; ++c)
#pragma unroll
        for (int r = 0; r < 8; ++r) {
            int m = q0 + r + mloc;
            ctxbf[(size_t)m * (NH * HD) + h * HD + 16 * c + n] =
                f2bf(oacc[c][r] / lrow[r]);
        }
}

// ---------------------------------------------------------------------------
// Launch
// ---------------------------------------------------------------------------
extern "C" void kernel_launch(void* const* d_in, const int* in_sizes, int n_in,
                              void* d_out, int out_size, void* d_ws, size_t ws_size,
                              hipStream_t stream)
{
    const float*     hs  = (const float*)d_in[0];
    const float*     wq  = (const float*)d_in[1];
    const float*     wk  = (const float*)d_in[2];
    const float*     wv  = (const float*)d_in[3];
    const float*     wo  = (const float*)d_in[4];
    const long long* pos = (const long long*)d_in[5];
    // d_in[6] (attention_mask) is causal; applied analytically in attn_kernel.
    float* out = (float*)d_out;

    // ---- workspace layout
    char* p = (char*)d_ws;
    float* qbuf = (float*)p;            p += (size_t)S_LEN * NH * HD * 4;
    float* kbuf = (float*)p;            p += (size_t)S_LEN * NKV * HD * 4;
    float* vbuf = (float*)p;            p += (size_t)S_LEN * NKV * HD * 4;
    float* kscale = (float*)p;          p += NKV * HD * 4;
    float* kbias  = (float*)p;          p += NKV * HD * 4;
    unsigned short* hsbf = (unsigned short*)p; p += (size_t)S_LEN * HDIM * 2;
    unsigned short* wqT  = (unsigned short*)p; p += (size_t)HDIM * NH * HD * 2;
    unsigned short* wkT  = (unsigned short*)p; p += (size_t)HDIM * NKV * HD * 2;
    unsigned short* wvT  = (unsigned short*)p; p += (size_t)HDIM * NKV * HD * 2;
    unsigned short* woT  = (unsigned short*)p; p += (size_t)NH * HD * HDIM * 2;
    unsigned short* qbf  = (unsigned short*)p; p += (size_t)NH * S_LEN * HD * 2;
    unsigned short* kbf  = (unsigned short*)p; p += (size_t)NKV * S_LEN * HD * 2;
    unsigned short* vbfT = (unsigned short*)p; p += (size_t)NKV * HD * S_LEN * 2;
    unsigned short* ctxbf = (unsigned short*)p;

    // 0) One-time (per launch) bf16 conversions / weight transposes
    {
        int n2 = S_LEN * HDIM / 2;
        conv_bf16_kernel<<<(n2 + 255) / 256, 256, 0, stream>>>(hs, hsbf, n2);
    }
    convT_kernel<<<dim3((NH * HD) / 32, HDIM / 32), 256, 0, stream>>>(wq, wqT, HDIM, NH * HD);
    convT_kernel<<<dim3((NKV * HD) / 32, HDIM / 32), 256, 0, stream>>>(wk, wkT, HDIM, NKV * HD);
    convT_kernel<<<dim3((NKV * HD) / 32, HDIM / 32), 256, 0, stream>>>(wv, wvT, HDIM, NKV * HD);
    convT_kernel<<<dim3(HDIM / 32, (NH * HD) / 32), 256, 0, stream>>>(wo, woT, NH * HD, HDIM);

    // 1) Projections (bf16 WMMA GEMMs, tiles 64x128, double-buffered)
    gemm_bf16_kernel<<<dim3((NH * HD) / 128, (S_LEN + 63) / 64), 256, 0, stream>>>(
        hsbf, wqT, qbuf, S_LEN, NH * HD, HDIM);
    gemm_bf16_kernel<<<dim3((NKV * HD) / 128, (S_LEN + 63) / 64), 256, 0, stream>>>(
        hsbf, wkT, kbuf, S_LEN, NKV * HD, HDIM);
    gemm_bf16_kernel<<<dim3((NKV * HD) / 128, (S_LEN + 63) / 64), 256, 0, stream>>>(
        hsbf, wvT, vbuf, S_LEN, NKV * HD, HDIM);

    // 2) RoPE on Q and K (fp32, in place)
    {
        int nq = S_LEN * NH * 64;
        rope_kernel<<<(nq + 255) / 256, 256, 0, stream>>>(qbuf, pos, NH, nq);
        int nk = S_LEN * NKV * 64;
        rope_kernel<<<(nk + 255) / 256, 256, 0, stream>>>(kbuf, pos, NKV, nk);
    }

    // 3) K channel stats (post-RoPE, full sequence)
    kstat_kernel<<<NKV * HD, 256, 0, stream>>>(kbuf, kscale, kbias);

    // 4) 4-bit group-64 quant-dequant of first QS_LEN tokens (in place)
    {
        int threads = QS_LEN * NKV * 2 * 32;
        quantk_kernel<<<(threads + 255) / 256, 256, 0, stream>>>(kbuf, kscale, kbias);
        quantv_kernel<<<(threads + 255) / 256, 256, 0, stream>>>(vbuf);
    }

    // 5) Convert Q/K/V to attention-friendly bf16 layouts
    {
        int tq = NH * S_LEN * 64;
        conv_head_kernel<<<(tq + 255) / 256, 256, 0, stream>>>(qbuf, qbf, NH, tq);
        int tk = NKV * S_LEN * 64;
        conv_head_kernel<<<(tk + 255) / 256, 256, 0, stream>>>(kbuf, kbf, NKV, tk);
        int tv = NKV * HD * (S_LEN / 2);
        conv_vT_kernel<<<(tv + 255) / 256, 256, 0, stream>>>(vbuf, vbfT);
    }

    // 6) Flash attention (WMMA + async double-buffered K/V), bf16 ctx
    attn_kernel<<<dim3(S_LEN / 16 / 5, NH), 160, 0, stream>>>(qbf, kbf, vbfT, ctxbf);

    // 7) Output projection
    gemm_bf16_kernel<<<dim3(HDIM / 128, (S_LEN + 63) / 64), 256, 0, stream>>>(
        ctxbf, woT, out, S_LEN, HDIM, NH * HD);
}
